// HAN_4776003633226
// MI455X (gfx1250) — matrix operand order repaired and checked by reading, not compiled, not run on hardware
//
#include <hip/hip_runtime.h>
#include <hip/hip_bf16.h>

typedef __attribute__((ext_vector_type(16))) _Float16 v16h;
typedef __attribute__((ext_vector_type(8)))  float    v8f;
typedef unsigned int u32x4 __attribute__((ext_vector_type(4)));
typedef unsigned int u32x8 __attribute__((ext_vector_type(8)));

#define NNODES 50000
#define NEDGE  800000
#define NPAIRS 100000

// ---------------------------------------------------------------------------
// Tensor Data Mover: issue a 2D tile load (nrows x rowlen fp32, row stride
// rowlen) from global memory into LDS. Descriptor built in SGPRs per
// CDNA5 ISA (D# group0 128b + group1 256b), issued via
// inline asm (portable across ROCm7.2 / amdgpu-toolchain builtin arities).
// Caller must be wave-uniform and follow with s_wait_tensorcnt + barrier.
// ---------------------------------------------------------------------------
__device__ __forceinline__ void tdm_issue(const void* gptr, unsigned int lds_addr,
                                          unsigned int rowlen, unsigned int nrows,
                                          unsigned int rows_total)
{
    const unsigned long long ga = (unsigned long long)gptr;
    u32x4 g0;
    g0[0] = 1u;                                          // count=1 (valid), load, user mode
    g0[1] = lds_addr;                                    // lds_addr[31:0]
    g0[2] = (unsigned int)ga;                            // global_addr[31:0]
    g0[3] = (unsigned int)((ga >> 32) & 0x01ffffffu)     // global_addr[56:32]
          | (2u << 30);                                  // type=2 ("image")
    u32x8 g1;
    g1[0] = 2u << 16;                                    // data_size=2 -> 4 bytes
    g1[1] = (rowlen & 0xffffu) << 16;                    // tensor_dim0[15:0]
    g1[2] = ((rowlen >> 16) & 0xffffu)                   // tensor_dim0[31:16]
          | ((rows_total & 0xffffu) << 16);              // tensor_dim1[15:0]
    g1[3] = ((rows_total >> 16) & 0xffffu)               // tensor_dim1[31:16]
          | ((rowlen & 0xffffu) << 16);                  // tile_dim0
    g1[4] = (nrows & 0xffffu);                           // tile_dim1 (tile_dim2=0)
    g1[5] = rowlen;                                      // tensor_dim0_stride[31:0]
    g1[6] = 0u;
    g1[7] = 0u;
    asm volatile("tensor_load_to_lds %0, %1" :: "s"(g0), "s"(g1) : "memory");
}

// ---------------------------------------------------------------------------
// One K=32 WMMA step: A fragment converted from LDS-resident fp32 rows
// (lane: row=lane&15, K groups {hi*8..hi*8+7, 16+hi*8..}), B fragment loaded
// as a single contiguous v16h from the pre-packed f16 weight image.
// ---------------------------------------------------------------------------
__device__ __forceinline__ v8f wmma_step(const float* ap, const _Float16* bp, v8f acc)
{
    v16h a;
#pragma unroll
    for (int e = 0; e < 8; ++e) {
        a[e]     = (_Float16)ap[e];
        a[8 + e] = (_Float16)ap[16 + e];
    }
    const v16h b = *(const v16h*)bp;
    return __builtin_amdgcn_wmma_f32_16x16x32_f16(false, a, false, b, (short)0,
                                                  acc, false, false);
}

// C tile store: VGPR r holds M = r + 8*(lane>=16), col = 16*warp + (lane&15).
__device__ __forceinline__ void wmma_epilogue(v8f acc, const float* bias,
                                              float* C, int mbase, int ldc)
{
    const int lane = threadIdx.x & 31;
    const int warp = threadIdx.x >> 5;
    const int hi   = lane >> 4;
    const int l15  = lane & 15;
    const int col  = warp * 16 + l15;
    const float bv = bias[col];
#pragma unroll
    for (int r = 0; r < 8; ++r) {
        float v = acc[r] + bv;
        v = v > 0.0f ? v : (__expf(v) - 1.0f);           // ELU
        C[(size_t)(mbase + r + hi * 8) * ldc + col] = v;
    }
}

// Pre-pack fp32 weight B[K,128] into the per-lane WMMA f16 fragment image:
// P[((ntile*(K/32) + kb)*32 + lane)*16 + e] = B[(kb*32 + (e>>3)*16 + (lane>>4)*8
//                                              + (e&7))*128 + ntile*16 + (lane&15)]
__global__ void packB_kernel(const float* __restrict__ B,
                             _Float16* __restrict__ P, int kshift)
{
    const int t = blockIdx.x * blockDim.x + threadIdx.x;
    const int K = 1 << kshift;
    if (t >= 128 * K) return;
    const int e    = t & 15;
    const int lane = (t >> 4) & 31;
    const int kb   = (t >> 9) & ((K >> 5) - 1);
    const int nt   = (t >> (kshift + 4)) & 7;
    const int col  = nt * 16 + (lane & 15);
    const int k    = kb * 32 + (e >> 3) * 16 + (lane >> 4) * 8 + (e & 7);
    P[t] = (_Float16)B[(size_t)k * 128 + col];
}

// C[M,128] = elu(A[M,128] @ B + bias); A tile staged via TDM; 8 waves/block.
__global__ void __launch_bounds__(256) gemm128_tdm_kernel(
    const float* __restrict__ A, const _Float16* __restrict__ pB,
    const float* __restrict__ bias, float* __restrict__ C, int ldc, int Mrows)
{
    __shared__ float sAf[16 * 128];
    const int mbase = blockIdx.x * 16;
    if (threadIdx.x < 32) {
        tdm_issue(A + (size_t)mbase * 128, (unsigned int)(uintptr_t)sAf,
                  128u, 16u, (unsigned int)Mrows);
        __builtin_amdgcn_s_wait_tensorcnt(0);
    }
    __syncthreads();
    const int lane = threadIdx.x & 31;
    const int warp = threadIdx.x >> 5;
    const int hi   = lane >> 4;
    const int l15  = lane & 15;
    v8f acc = {};
#pragma unroll
    for (int kb = 0; kb < 4; ++kb) {
        const float*    ap = sAf + l15 * 128 + kb * 32 + hi * 8;
        const _Float16* bp = pB + ((size_t)(warp * 4 + kb) * 32 + lane) * 16;
        acc = wmma_step(ap, bp, acc);
    }
    wmma_epilogue(acc, bias, C, mbase, ldc);
}

// C[M,128] = elu(concat(Aa,Ab)[M,256] @ B + bias); two TDM tiles per block.
__global__ void __launch_bounds__(256) gemm_concat_tdm_kernel(
    const float* __restrict__ Aa, const float* __restrict__ Ab,
    const _Float16* __restrict__ pB, const float* __restrict__ bias,
    float* __restrict__ C, int Mrows)
{
    __shared__ float sAf[2 * 16 * 128];
    const int mbase = blockIdx.x * 16;
    if (threadIdx.x < 32) {
        tdm_issue(Aa + (size_t)mbase * 128, (unsigned int)(uintptr_t)sAf,
                  128u, 16u, (unsigned int)Mrows);
        tdm_issue(Ab + (size_t)mbase * 128,
                  (unsigned int)(uintptr_t)(sAf + 2048), 128u, 16u,
                  (unsigned int)Mrows);
        __builtin_amdgcn_s_wait_tensorcnt(0);
    }
    __syncthreads();
    const int lane = threadIdx.x & 31;
    const int warp = threadIdx.x >> 5;
    const int hi   = lane >> 4;
    const int l15  = lane & 15;
    v8f acc = {};
#pragma unroll
    for (int kb = 0; kb < 8; ++kb) {
        const int tile = kb >> 2;
        const int kk   = (kb & 3) * 32;
        const float*    ap = sAf + tile * 2048 + l15 * 128 + kk + hi * 8;
        const _Float16* bp = pB + ((size_t)(warp * 8 + kb) * 32 + lane) * 16;
        acc = wmma_step(ap, bp, acc);
    }
    wmma_epilogue(acc, bias, C, mbase, 128);
}

// ---------------------------------------------------------------------------
// r_vec: per edge type, emb = elu(edge_feat @ fc_edge) [128]; normalize pairs.
// ---------------------------------------------------------------------------
__global__ void rvec_kernel(const float* __restrict__ edge_feat,
                            const float* __restrict__ fc_edge,
                            float* __restrict__ r)
{
    __shared__ float emb[128];
    const int et = blockIdx.x;
    const int d  = threadIdx.x;
    float acc = 0.0f;
    for (int k = 0; k < 128; ++k)
        acc += edge_feat[et * 128 + k] * fc_edge[((size_t)et * 128 + k) * 128 + d];
    acc = acc > 0.0f ? acc : (__expf(acc) - 1.0f);
    emb[d] = acc;
    __syncthreads();
    if (d < 64) {
        const float a = emb[d], b = emb[64 + d];
        float nrm = sqrtf(a * a + b * b);
        nrm = fmaxf(nrm, 1e-12f);
        r[(et * 64 + d) * 2 + 0] = a / nrm;
        r[(et * 64 + d) * 2 + 1] = b / nrm;
    }
}

// ---------------------------------------------------------------------------
// Complex rotation per path; writes BOTH h_new and agg (agg is the identity
// term of (h_new + segment_sum); scatter then accumulates on top of it).
// ---------------------------------------------------------------------------
__global__ void rotate_kernel(const float* __restrict__ emb,
                              const float* __restrict__ r,
                              float* __restrict__ out_h,
                              float* __restrict__ out_agg,
                              int e0, int e1, int ne, int is_in, int n_nodes)
{
    const long t = (long)blockIdx.x * blockDim.x + threadIdx.x;
    if (t >= (long)n_nodes * 64) return;
    const int n = (int)(t >> 6);
    const int j = (int)(t & 63);
    const float2 p = ((const float2*)emb)[(size_t)n * 64 + j];
    float t1 = p.x, t2 = p.y;
    int edges[2] = {e0, e1};
    for (int q = 0; q < ne; ++q) {
        const int e = edges[q];
        const float r0 = r[((e - 1) * 64 + j) * 2 + 0];
        const float r1 = r[((e - 1) * 64 + j) * 2 + 1];
        if (is_in) {  // reference's sequential-assignment semantics
            const float nt1 = t1 * r0 - t2 * r1;
            t2 = nt1 * r1 + t2 * r0;
            t1 = nt1;
        } else {
            const float nt1 = t1 * r0 + t2 * r1;
            t2 = t2 * r0 - nt1 * r1;
            t1 = nt1;
        }
    }
    out_h[(size_t)n * 128 + j]        = t1;
    out_h[(size_t)n * 128 + 64 + j]   = t2;
    out_agg[(size_t)n * 128 + j]      = t1;
    out_agg[(size_t)n * 128 + 64 + j] = t2;
}

// ---------------------------------------------------------------------------
// Segment-sum: AGG[dst[e]] += H[src[e]]; thread = (edge, 4 consecutive feats).
// Dominant memory op: ~820 MB per call, fp32 atomics land in the 192MB L2.
// ---------------------------------------------------------------------------
__global__ void scatter_kernel(const float* __restrict__ H,
                               const int* __restrict__ src,
                               const int* __restrict__ dst,
                               float* __restrict__ AGG, int nE)
{
    const long t = (long)blockIdx.x * blockDim.x + threadIdx.x;
    if (t >= (long)nE * 32) return;
    const int e = (int)(t >> 5);
    const int f = (int)((t & 31) << 2);
    const int s = src[e], d = dst[e];
    const float4 v = *reinterpret_cast<const float4*>(&H[(size_t)s * 128 + f]);
    float* p = &AGG[(size_t)d * 128 + f];
    __hip_atomic_fetch_add(p + 0, v.x, __ATOMIC_RELAXED, __HIP_MEMORY_SCOPE_AGENT);
    __hip_atomic_fetch_add(p + 1, v.y, __ATOMIC_RELAXED, __HIP_MEMORY_SCOPE_AGENT);
    __hip_atomic_fetch_add(p + 2, v.z, __ATOMIC_RELAXED, __HIP_MEMORY_SCOPE_AGENT);
    __hip_atomic_fetch_add(p + 3, v.w, __ATOMIC_RELAXED, __HIP_MEMORY_SCOPE_AGENT);
}

__global__ void zero_kernel(float* __restrict__ p, long n)
{
    const long t = (long)blockIdx.x * blockDim.x + threadIdx.x;
    if (t < n) p[t] = 0.0f;
}

// ---------------------------------------------------------------------------
// Semantic attention logits: one wave per z-row (rows = 3N, row = n*3 + path).
// ---------------------------------------------------------------------------
__global__ void sem_kernel(const float* __restrict__ z,
                           const float* __restrict__ sw1,  // [128,64]
                           const float* __restrict__ sb1,  // [64]
                           const float* __restrict__ sw2,  // [64]
                           float* __restrict__ sums, int rows)
{
    const long t = (long)blockIdx.x * blockDim.x + threadIdx.x;
    const int row  = (int)(t >> 5);
    const int lane = (int)(t & 31);
    if (row >= rows) return;
    const float* zr = z + (size_t)row * 128;
    float partial = 0.0f;
    for (int h = lane; h < 64; h += 32) {
        float acc = sb1[h];
        for (int k = 0; k < 128; ++k) acc += zr[k] * sw1[(size_t)k * 64 + h];
        partial += tanhf(acc) * sw2[h];
    }
    for (int off = 16; off > 0; off >>= 1)
        partial += __shfl_xor(partial, off, 32);
    if (lane == 0)
        __hip_atomic_fetch_add(&sums[row % 3], partial,
                               __ATOMIC_RELAXED, __HIP_MEMORY_SCOPE_AGENT);
}

__global__ void beta_kernel(const float* __restrict__ sums,
                            float* __restrict__ beta, float invN)
{
    float m[3];
    for (int i = 0; i < 3; ++i) m[i] = sums[i] * invN;
    const float mx = fmaxf(m[0], fmaxf(m[1], m[2]));
    float s = 0.0f, e[3];
    for (int i = 0; i < 3; ++i) { e[i] = __expf(m[i] - mx); s += e[i]; }
    for (int i = 0; i < 3; ++i) beta[i] = e[i] / s;
}

__global__ void wsum_kernel(const float* __restrict__ z,
                            const float* __restrict__ beta,
                            float* __restrict__ out, int n_nodes)
{
    const long t = (long)blockIdx.x * blockDim.x + threadIdx.x;
    if (t >= (long)n_nodes * 128) return;
    const int n = (int)(t >> 7);
    const int d = (int)(t & 127);
    const size_t base = (size_t)n * 384 + d;
    out[t] = z[base] * beta[0] + z[base + 128] * beta[1] + z[base + 256] * beta[2];
}

__global__ void predict_kernel(const float* __restrict__ emb,
                               const int* __restrict__ eidx,
                               const float* __restrict__ pw,
                               float* __restrict__ logits, int npair)
{
    const long t = (long)blockIdx.x * blockDim.x + threadIdx.x;
    if (t >= (long)npair * 2) return;
    const int p = (int)(t >> 1);
    const int o = (int)(t & 1);
    const int i0 = eidx[p * 2 + 0];
    const int i1 = eidx[p * 2 + 1];
    float acc = 0.0f;
    for (int k = 0; k < 128; ++k) acc += emb[(size_t)i0 * 128 + k] * pw[k * 2 + o];
    for (int k = 0; k < 128; ++k) acc += emb[(size_t)i1 * 128 + k] * pw[(128 + k) * 2 + o];
    logits[(size_t)p * 2 + o] = acc;
}

// ---------------------------------------------------------------------------
extern "C" void kernel_launch(void* const* d_in, const int* in_sizes, int n_in,
                              void* d_out, int out_size, void* d_ws, size_t ws_size,
                              hipStream_t stream)
{
    const float* node_feat    = (const float*)d_in[0];
    const float* fc_node_w    = (const float*)d_in[1];
    const float* fc_node_b    = (const float*)d_in[2];
    const float* edge_feat_in = (const float*)d_in[3];
    const float* fc_edge_in   = (const float*)d_in[4];
    const float* edge_feat_out= (const float*)d_in[5];
    const float* fc_edge_out  = (const float*)d_in[6];
    const float* gin_w_in     = (const float*)d_in[7];
    const float* gin_b_in     = (const float*)d_in[8];
    const float* sem_w1_in    = (const float*)d_in[9];
    const float* sem_b1_in    = (const float*)d_in[10];
    const float* sem_w2_in    = (const float*)d_in[11];
    const float* gin_w_out    = (const float*)d_in[12];
    const float* gin_b_out    = (const float*)d_in[13];
    const float* sem_w1_out   = (const float*)d_in[14];
    const float* sem_b1_out   = (const float*)d_in[15];
    const float* sem_w2_out   = (const float*)d_in[16];
    const float* fc_w         = (const float*)d_in[17];
    const float* fc_b         = (const float*)d_in[18];
    const float* predict_w    = (const float*)d_in[19];
    const int*   src_in       = (const int*)d_in[20];
    const int*   dst_in       = (const int*)d_in[21];
    const int*   src_out      = (const int*)d_in[22];
    const int*   dst_out      = (const int*)d_in[23];
    const int*   edge_indices = (const int*)d_in[24];

    const int N = NNODES, E = NEDGE, NP = NPAIRS, Lnum = 2;

    // Workspace layout (floats): h_new|agg|z|e_in|e_out|small|packed-f16 B
    float* ws     = (float*)d_ws;
    float* h_new  = ws;
    float* agg    = ws + (size_t)128 * N;
    float* z      = ws + (size_t)256 * N;   // [N,3,128] == [3N,128]
    float* e_inb  = ws + (size_t)640 * N;
    float* e_outb = ws + (size_t)768 * N;
    float* small  = ws + (size_t)896 * N;
    float* r_in   = small;                  // [2][64][2]
    float* r_out  = small + 256;
    float* sums   = small + 512;            // [3]
    float* beta   = small + 520;            // [3]
    _Float16* pB      = (_Float16*)(small + 1024);  // packed weight images
    _Float16* pB_node = pB;                         // 16384
    _Float16* pB_gin  = pB + 16384;                 // 12 * 16384
    _Float16* pB_fc   = pB + (size_t)16384 * 13;    // 2 * 32768

    float* node_emb = (float*)d_out;                 // [N,128] final output slot
    float* logits   = node_emb + (size_t)N * 128;    // [NPAIR,2]

    const int g_rot  = (N * 64 + 255) / 256;
    const int g_sc   = (int)(((long)E * 32 + 255) / 256);
    const int g_sem  = (int)(((long)3 * N * 32 + 255) / 256);
    const int g_wsum = (N * 128 + 255) / 256;
    const int g_gemm = N / 16;   // 3125, exact

    // Pre-pack all weight matrices into WMMA f16 fragment images.
    packB_kernel<<<64, 256, 0, stream>>>(fc_node_w, pB_node, 7);
    for (int tag = 0; tag < 2; ++tag) {
        const float* gw = tag == 0 ? gin_w_in : gin_w_out;
        for (int q = 0; q < 6; ++q)   // q = l*3+i
            packB_kernel<<<64, 256, 0, stream>>>(
                gw + (size_t)q * 16384, pB_gin + (size_t)(tag * 6 + q) * 16384, 7);
    }
    for (int l = 0; l < Lnum; ++l)
        packB_kernel<<<128, 256, 0, stream>>>(
            fc_w + (size_t)l * 32768, pB_fc + (size_t)l * 32768, 8);

    // Edge sequences per path: in = path order, out = reversed
    const int in_e0[3]  = {1, 2, 1}, in_e1[3]  = {0, 0, 2}, in_ne[3]  = {1, 1, 2};
    const int out_e0[3] = {1, 2, 2}, out_e1[3] = {0, 0, 1}, out_ne[3] = {1, 1, 2};

    rvec_kernel<<<2, 128, 0, stream>>>(edge_feat_in,  fc_edge_in,  r_in);
    rvec_kernel<<<2, 128, 0, stream>>>(edge_feat_out, fc_edge_out, r_out);

    // node_emb = elu(node_feat @ fc_node_w + b)
    gemm128_tdm_kernel<<<g_gemm, 256, 0, stream>>>(node_feat, pB_node,
                                                   fc_node_b, node_emb, 128, N);

    for (int l = 0; l < Lnum; ++l) {
        for (int tag = 0; tag < 2; ++tag) {
            const int is_in = (tag == 0);
            const float* rv   = is_in ? r_in : r_out;
            const int* srcP   = is_in ? src_in : src_out;
            const int* dstP   = is_in ? dst_in : dst_out;
            const float* ginB = is_in ? gin_b_in : gin_b_out;
            const float* sw1  = is_in ? sem_w1_in : sem_w1_out;
            const float* sb1  = is_in ? sem_b1_in : sem_b1_out;
            const float* sw2  = is_in ? sem_w2_in : sem_w2_out;
            const int* pe0    = is_in ? in_e0 : out_e0;
            const int* pe1    = is_in ? in_e1 : out_e1;
            const int* pne    = is_in ? in_ne : out_ne;
            float* e_dst      = is_in ? e_inb : e_outb;

            for (int i = 0; i < 3; ++i) {
                rotate_kernel<<<g_rot, 256, 0, stream>>>(
                    node_emb, rv, h_new, agg, pe0[i], pe1[i], pne[i], is_in, N);
                scatter_kernel<<<g_sc, 256, 0, stream>>>(
                    h_new, srcP + (size_t)i * E, dstP + (size_t)i * E, agg, E);
                // z[:, i, :] = elu((h_new + segsum) @ gin_w[l,i] + gin_b[l,i])
                gemm128_tdm_kernel<<<g_gemm, 256, 0, stream>>>(
                    agg, pB_gin + (size_t)(tag * 6 + l * 3 + i) * 16384,
                    ginB + (size_t)(l * 3 + i) * 128,
                    z + (size_t)i * 128, 384, N);
            }
            zero_kernel<<<1, 32, 0, stream>>>(sums, 4);
            sem_kernel<<<g_sem, 256, 0, stream>>>(
                z, sw1 + (size_t)l * 128 * 64, sb1 + (size_t)l * 64,
                sw2 + (size_t)l * 64, sums, 3 * N);
            beta_kernel<<<1, 1, 0, stream>>>(sums, beta, 1.0f / (float)N);
            wsum_kernel<<<g_wsum, 256, 0, stream>>>(z, beta, e_dst, N);
        }
        // node_emb = elu(concat(e_in, e_out) @ fc_w[l] + fc_b[l])
        gemm_concat_tdm_kernel<<<g_gemm, 256, 0, stream>>>(
            e_inb, e_outb, pB_fc + (size_t)l * 32768,
            fc_b + (size_t)l * 128, node_emb, N);
    }

    predict_kernel<<<(NP * 2 + 255) / 256, 256, 0, stream>>>(
        node_emb, edge_indices, predict_w, logits, NP);
}
